// Linear4bit_63350767616593
// MI455X (gfx1250) — matrix operand order repaired and checked by reading, not compiled, not run on hardware
//
#include <hip/hip_runtime.h>
#include <hip/hip_bf16.h>

typedef __attribute__((ext_vector_type(16))) _Float16 v16h;
typedef __attribute__((ext_vector_type(8)))  _Float16 v8h;
typedef __attribute__((ext_vector_type(2)))  _Float16 v2h;
typedef __attribute__((ext_vector_type(8)))  float    v8f;
typedef __attribute__((ext_vector_type(4)))  float    v4f;
typedef __attribute__((ext_vector_type(4)))  int      v4i;

#define OUT_F   11008
#define IN_F    4096
#define KSTEP   64       // K per iteration (2 WMMA K-chunks of 32)
#define NBLK    64       // output columns per block -> 172 blocks
#define LDS_STR 72       // padded f16 row stride (144B, keeps ds_b128 16B-aligned)

__constant__ float NF4_TAB[16] = {
    -1.0f, -0.6961928009986877f, -0.5250730514526367f, -0.39491748809814453f,
    -0.28444138169288635f, -0.18477343022823334f, -0.09105003625154495f, 0.0f,
    0.07958029955625534f, 0.16093020141124725f, 0.24611230194568634f,
    0.33791524171829224f, 0.44070982933044434f, 0.5626170039176941f,
    0.7229568362236023f, 1.0f};

__global__ __launch_bounds__(256)
void nf4_linear_wmma_kernel(const float* __restrict__ x,
                            const int*   __restrict__ wpk,
                            const float* __restrict__ absmax,
                            float*       __restrict__ out) {
    // double-buffered activation tile (64 rows x 64 K as f16)
    __shared__ _Float16 ldsA[2][64 * LDS_STR];
    // byte -> {hi-nibble value, lo-nibble value} pair table (1 KB)
    __shared__ v2h cb2[256];

    const int tid   = threadIdx.x;
    const int lane  = tid & 31;
    const int wave  = tid >> 5;
    const int l16   = lane & 15;   // N within tile / M within half
    const int lh    = lane >> 4;   // lane-half -> K group
    const int ntile = wave & 3;    // 4 N strips of 16
    const int mhalf = wave >> 2;   // 2 waves split M (2 tiles each)

    {   // one table entry per thread
        v2h e;
        e[0] = (_Float16)NF4_TAB[(tid >> 4) & 15];   // element 2j   (hi nibble)
        e[1] = (_Float16)NF4_TAB[tid & 15];          // element 2j+1 (lo nibble)
        cb2[tid] = e;
    }

    const int n    = blockIdx.x * NBLK + ntile * 16 + l16;  // 172*64 == 11008 exact
    const int prow = n * (IN_F / 2);     // packed-int32 row base
    const int srow = n * (IN_F / 64);    // absmax row base

    // cooperative x staging: 4 threads per row, 16 floats each
    const int arow  = tid >> 2;
    const int akoff = (tid & 3) * 16;
    const float* xrow = x + arow * IN_F + akoff;
    const int*   wpl  = wpk + prow + lh * 8;   // lane's packed-weight base

    auto cvtStore = [&](int buf, v4f f0, v4f f1, v4f f2, v4f f3) {
        v8h h0, h1;
#pragma unroll
        for (int j = 0; j < 4; ++j) {
            h0[j] = (_Float16)f0[j]; h0[4 + j] = (_Float16)f1[j];
            h1[j] = (_Float16)f2[j]; h1[4 + j] = (_Float16)f3[j];
        }
        _Float16* dst = &ldsA[buf][arow * LDS_STR + akoff];
        *(v8h*)dst       = h0;
        *(v8h*)(dst + 8) = h1;
    };

    v8f acc0 = {}, acc1 = {};

    // decode prefetched packed weights into two scaled B fragments
    auto decode = [&](v4i W0, v4i W1, v4i W2, v4i W3, float sc, v16h& B0, v16h& B1) {
        const _Float16 hsc = (_Float16)sc;
        v2h hs2; hs2[0] = hsc; hs2[1] = hsc;
        v2h* q0 = (v2h*)&B0;
        v2h* q1 = (v2h*)&B1;
#pragma unroll
        for (int j = 0; j < 4; ++j) {
            q0[j]     = cb2[W0[j]] * hs2;   // K rel  0..7
            q0[4 + j] = cb2[W1[j]] * hs2;   // K rel  8..15
            q1[j]     = cb2[W2[j]] * hs2;   // K rel 32..39
            q1[4 + j] = cb2[W3[j]] * hs2;   // K rel 40..47
        }
    };

    // A fragments (2 M tiles x 2 K-chunks) + 4 WMMAs
    auto matmul = [&](int p, const v16h& B0, const v16h& B1) {
        v16h af[2][2];
#pragma unroll
        for (int a = 0; a < 2; ++a) {
            const int m = (mhalf * 2 + a) * 16 + l16;
            const _Float16* base = &ldsA[p][m * LDS_STR + lh * 16];
#pragma unroll
            for (int s = 0; s < 2; ++s) {
                *(v8h*)&af[a][s]       = *(const v8h*)(base + s * 32);
                *((v8h*)&af[a][s] + 1) = *(const v8h*)(base + s * 32 + 8);
            }
        }
        acc0 = __builtin_amdgcn_wmma_f32_16x16x32_f16(false, af[0][0], false, B0,
                                                      (short)0, acc0, false, false);
        acc0 = __builtin_amdgcn_wmma_f32_16x16x32_f16(false, af[0][1], false, B1,
                                                      (short)0, acc0, false, false);
        acc1 = __builtin_amdgcn_wmma_f32_16x16x32_f16(false, af[1][0], false, B0,
                                                      (short)0, acc1, false, false);
        acc1 = __builtin_amdgcn_wmma_f32_16x16x32_f16(false, af[1][1], false, B1,
                                                      (short)0, acc1, false, false);
    };

    // ---- prologue: stage kb=0 tile into buffer 0; prefetch kb=0 weights ----
    {
        v4f f0 = *(const v4f*)(xrow);
        v4f f1 = *(const v4f*)(xrow + 4);
        v4f f2 = *(const v4f*)(xrow + 8);
        v4f f3 = *(const v4f*)(xrow + 12);
        cvtStore(0, f0, f1, f2, f3);
    }
    __syncthreads();

    v4i w0 = __builtin_nontemporal_load((const v4i*)wpl);
    v4i w1 = __builtin_nontemporal_load((const v4i*)(wpl + 4));
    v4i w2 = __builtin_nontemporal_load((const v4i*)(wpl + 16));
    v4i w3 = __builtin_nontemporal_load((const v4i*)(wpl + 20));
    float sc = absmax[srow];

    // steady state: weights/scale prefetched one iteration ahead (register
    // double-buffer); x tile loads issued at top, converted+stored after WMMAs.
#pragma unroll 2
    for (int kb = 0; kb < IN_F - KSTEP; kb += KSTEP) {
        const int p = (kb >> 6) & 1;

        // prefetch next iteration's single-use weight stream (non-temporal)
        const int* wpn = wpl + ((kb + KSTEP) >> 1);
        v4i nw0 = __builtin_nontemporal_load((const v4i*)wpn);
        v4i nw1 = __builtin_nontemporal_load((const v4i*)(wpn + 4));
        v4i nw2 = __builtin_nontemporal_load((const v4i*)(wpn + 16));
        v4i nw3 = __builtin_nontemporal_load((const v4i*)(wpn + 20));
        const float nsc = absmax[srow + ((kb + KSTEP) >> 6)];

        // next x tile: loads now, convert+store after compute
        const float* xs = xrow + kb + KSTEP;
        v4f f0 = *(const v4f*)(xs);
        v4f f1 = *(const v4f*)(xs + 4);
        v4f f2 = *(const v4f*)(xs + 8);
        v4f f3 = *(const v4f*)(xs + 12);

        // compute with weights prefetched last iteration (no fresh vmem wait)
        v16h b0, b1;
        decode(w0, w1, w2, w3, sc, b0, b1);
        matmul(p, b0, b1);

        // publish next x tile
        cvtStore(1 - p, f0, f1, f2, f3);

        // rotate prefetch registers (renamed away by unroll)
        w0 = nw0; w1 = nw1; w2 = nw2; w3 = nw3; sc = nsc;
        __syncthreads();
    }

    // peeled last iteration: no prefetch, no staging, no barrier
    {
        v16h b0, b1;
        decode(w0, w1, w2, w3, sc, b0, b1);
        matmul(((IN_F - KSTEP) >> 6) & 1, b0, b1);
    }

    // ---- store D (non-temporal, single-use output) ----
#pragma unroll
    for (int a = 0; a < 2; ++a) {
        const v8f accv = a ? acc1 : acc0;
#pragma unroll
        for (int r = 0; r < 8; ++r) {
            const int row = (mhalf * 2 + a) * 16 + r + 8 * lh;
            __builtin_nontemporal_store(accv[r], &out[row * OUT_F + n]);
        }
    }
}

extern "C" void kernel_launch(void* const* d_in, const int* in_sizes, int n_in,
                              void* d_out, int out_size, void* d_ws, size_t ws_size,
                              hipStream_t stream) {
    (void)in_sizes; (void)n_in; (void)d_ws; (void)ws_size; (void)out_size;
    const float* x      = (const float*)d_in[0];
    const int*   wpk    = (const int*)d_in[1];
    const float* absmax = (const float*)d_in[2];
    float*       out    = (float*)d_out;

    dim3 grid(OUT_F / NBLK);   // 172 blocks, 8 waves each
    dim3 block(256);
    nf4_linear_wmma_kernel<<<grid, block, 0, stream>>>(x, wpk, absmax, out);
}
